// WarpNet_90821378441561
// MI455X (gfx1250) — matrix operand order repaired and checked by reading, not compiled, not run on hardware
//
#include <hip/hip_runtime.h>

// ---------------------------------------------------------------------------
// WarpNet (deformable conv) fused forward for gfx1250 (MI455X, wave32, WMMA).
//   kernel 1: weights fp32 -> bf16; comw padded to 256 rows (zero-filled),
//             com_b padded to 256 floats -> phase 1 is fully branch-free.
//   kernel 2: per 16-pixel row tile:
//     stage   : full 16x576 im2col tile of `offsets` -> LDS (bf16), 1 barrier
//     phase 1 : offset conv GEMM 16x576x(216->256) via v_wmma_f32_16x16x32_bf16
//               (8 waves x 2 N-tiles, barrier-free pipelined K loop)
//     phase 3 : bilinear sampling * sigmoid(mask), clamped unconditional taps,
//               validity folded into weights -> bf16 val tile (reuses A LDS)
//     phase 4 : output GEMM 16x576x64, N-tile x K-half per wave, LDS reduce
// Fragment loads: B = 2x global_load_b128, A = 2x ds_load_b128 (k = hl*16 + s,
// same order in A and B); LDS tile row stride 584 (73 16B segments, odd).
// ---------------------------------------------------------------------------

typedef __attribute__((ext_vector_type(16))) __bf16 v16bf;
typedef __attribute__((ext_vector_type(8)))  __bf16 v8bf;
typedef __attribute__((ext_vector_type(8)))  float  v8f;

constexpr int kB   = 4;
constexpr int kC   = 64;
constexpr int kH   = 128;
constexpr int kW   = 128;
constexpr int kO   = 64;
constexpr int kKd  = 576;      // C * 9 reduction dim
constexpr int kNom = 216;      // 3 * DG * 9 offset-conv channels
constexpr int kNomPad = 256;   // padded so both wave N-tiles are in-bounds
constexpr int kKSteps = 18;    // 576 / 32
constexpr int kRow    = 584;   // LDS tile row stride (73 x 16B segments)

__device__ __forceinline__ __bf16 f2bf(float f) {
  unsigned u = __builtin_bit_cast(unsigned, f);
  u += 0x7FFFu + ((u >> 16) & 1u);              // round-to-nearest-even
  unsigned short s = (unsigned short)(u >> 16);
  return __builtin_bit_cast(__bf16, s);
}

// ---------------- weight fp32 -> bf16 conversion + padding ----------------
__global__ void wconv_kernel(const float* __restrict__ com_w,
                             const float* __restrict__ dc_w,
                             const float* __restrict__ com_b,
                             __bf16* __restrict__ comw,
                             __bf16* __restrict__ dcw,
                             float*  __restrict__ comb_pad) {
  int i = blockIdx.x * 256 + threadIdx.x;
  if (i < kNomPad * kKd) comw[i] = (i < kNom * kKd) ? f2bf(com_w[i]) : f2bf(0.0f);
  if (i < kO * kKd)      dcw[i]  = f2bf(dc_w[i]);
  if (i < kNomPad)       comb_pad[i] = (i < kNom) ? com_b[i] : 0.0f;
}

// ---------------- fused deformable conv -----------------------------------
__global__ __launch_bounds__(256)
void warpnet_fused(const float*  __restrict__ input,
                   const float*  __restrict__ offsets,
                   const __bf16* __restrict__ comw,
                   const float*  __restrict__ comb,
                   const __bf16* __restrict__ dcw,
                   const float*  __restrict__ dc_b,
                   float*        __restrict__ out) {
  // om (phases 1-3) and the phase-4 partial buffer are never live together.
  __shared__ union {
    float om[16][kNomPad];      // 16 KB
    float part[2][16][kO];      //  8 KB
  } s_u;
  // One bf16 tile, used twice: im2col A (stage+phase 1), then val (phase 3/4).
  __shared__ __align__(16) __bf16 s_t[16][kRow];   // 18688 B

  const int tid  = threadIdx.x;
  const int lane = tid & 31;
  const int wv   = tid >> 5;           // wave id 0..7
  const int hl   = lane >> 4;          // lane half (0/1)
  const int lm   = lane & 15;

  const int p0 = blockIdx.x * 16;      // 16 consecutive pixels (one row chunk)
  const int b  = p0 >> 14;
  const int y  = (p0 >> 7) & 127;
  const int x0 = p0 & 127;

  // Warm caches with the central 3 input rows of every channel (likely taps).
  if (tid < kC * 3) {
    int ch = tid / 3, r = tid - ch * 3;
    int yy = y - 1 + r;
    if (yy >= 0 && yy < kH)
      __builtin_prefetch(&input[((b * kC + ch) << 14) + (yy << 7) + x0], 0, 3);
  }

  // ----- stage the full 16x576 im2col A tile (offsets, fp32 -> bf16) -----
  for (int idx = tid; idx < 16 * kKd; idx += 256) {
    int m   = idx / kKd;
    int kk  = idx - m * kKd;           // reduction index: c*9 + tap
    int c   = kk / 9;
    int tap = kk - c * 9;
    int ky  = tap / 3, kx = tap - ky * 3;
    int yy  = y + ky - 1;
    int xx  = x0 + m + kx - 1;
    float v = 0.0f;
    if (yy >= 0 && yy < kH && xx >= 0 && xx < kW)
      v = offsets[((b * kC + c) << 14) + (yy << 7) + xx];
    s_t[m][kk] = f2bf(v);
  }
  __syncthreads();

  // ----- phase 1: offset conv GEMM, barrier-free pipelined K loop -----
  v8f acc0 = {};
  v8f acc1 = {};
  const int t0 = wv;                   // N-tiles for this wave (both always run;
  const int t1 = wv + 8;               //  rows >= 216 are zero-padded weights)

  for (int ks = 0; ks < kKSteps; ++ks) {
    // A fragment: two contiguous 16B runs -> 2x ds_load_b128
    const v8bf* ap = (const v8bf*)(&s_t[lm][ks * 32 + hl * 16]);
    v8bf alo = ap[0];                  // K = hl*16 .. hl*16+7
    v8bf ahi = ap[1];                  // K = hl*16+8 .. hl*16+15
    v16bf a = __builtin_shufflevector(alo, ahi, 0, 1, 2, 3, 4, 5, 6, 7,
                                      8, 9, 10, 11, 12, 13, 14, 15);
    // B fragments: 32 contiguous bytes per lane -> 2x global_load_b128 each
    const int kgb = ks * 32 + hl * 16;
    v16bf b0 = *(const v16bf*)(comw + (t0 * 16 + lm) * kKd + kgb);
    v16bf b1 = *(const v16bf*)(comw + (t1 * 16 + lm) * kKd + kgb);

    acc0 = __builtin_amdgcn_wmma_f32_16x16x32_bf16(false, a, false, b0,
                                                   (short)0, acc0, false, false);
    acc1 = __builtin_amdgcn_wmma_f32_16x16x32_bf16(false, a, false, b1,
                                                   (short)0, acc1, false, false);
  }

  // write om (+bias) to LDS; C layout: VGPR r -> M = r + hl*8, N = lm
#pragma unroll
  for (int r = 0; r < 8; ++r) {
    int m = r + hl * 8;
    s_u.om[m][t0 * 16 + lm] = acc0[r] + comb[t0 * 16 + lm];
    s_u.om[m][t1 * 16 + lm] = acc1[r] + comb[t1 * 16 + lm];
  }
  __syncthreads();   // om ready; all phase-1 reads of s_t are done

  // ----- phase 3: bilinear sampling * sigmoid(mask) -> bf16 val tile -----
  for (int pos = tid; pos < 16 * 72; pos += 256) {
    int m  = pos / 72;
    int gk = pos - m * 72;
    int g  = gk / 9;
    int k  = gk - g * 9;
    int ky = k / 3, kx = k - ky * 3;

    float dy = s_u.om[m][g * 18 + k * 2];
    float dx = s_u.om[m][g * 18 + k * 2 + 1];
    float mo = s_u.om[m][144 + g * 9 + k];
    float mv = 1.0f / (1.0f + __expf(-mo));

    float py = dy + (float)(y + ky - 1);
    float px = dx + (float)(x0 + m + kx - 1);
    float fy = floorf(py), fx = floorf(px);
    float wy = py - fy,    wx = px - fx;
    int yi = (int)fy, xi = (int)fx;

    // fold validity into the bilinear weights; clamp tap indices
    float w00 = (1.0f - wy) * (1.0f - wx);
    float w01 = (1.0f - wy) * wx;
    float w10 = wy * (1.0f - wx);
    float w11 = wy * wx;
    bool vy0 = (yi >= 0) && (yi < kH);
    bool vy1 = (yi + 1 >= 0) && (yi + 1 < kH);
    bool vx0 = (xi >= 0) && (xi < kW);
    bool vx1 = (xi + 1 >= 0) && (xi + 1 < kW);
    w00 = (vy0 && vx0) ? w00 : 0.0f;
    w01 = (vy0 && vx1) ? w01 : 0.0f;
    w10 = (vy1 && vx0) ? w10 : 0.0f;
    w11 = (vy1 && vx1) ? w11 : 0.0f;
    int yc0 = min(max(yi, 0), kH - 1),     yc1 = min(max(yi + 1, 0), kH - 1);
    int xc0 = min(max(xi, 0), kW - 1),     xc1 = min(max(xi + 1, 0), kW - 1);
    int o00 = (yc0 << 7) + xc0, o01 = (yc0 << 7) + xc1;
    int o10 = (yc1 << 7) + xc0, o11 = (yc1 << 7) + xc1;

#pragma unroll
    for (int cg = 0; cg < 8; ++cg) {
      int ch = g * 8 + cg;
      const float* base = input + ((b * kC + ch) << 14);
      float v = (w00 * base[o00] + w01 * base[o01] +
                 w10 * base[o10] + w11 * base[o11]) * mv;
      s_t[m][ch * 9 + k] = f2bf(v);    // kdim = (g*8+cg)*9 + k matches dc_w
    }
  }
  __syncthreads();

  // ----- phase 4: output GEMM 16x576x64; all 8 waves: N-tile x K-half -----
  {
    const int nt = wv & 3;             // output-channel tile (o = nt*16 + lm)
    const int kh = wv >> 2;            // K-half: ks in [kh*9, kh*9+9)
    v8f acc = {};
    for (int ks = kh * 9; ks < kh * 9 + 9; ++ks) {
      const v8bf* vp = (const v8bf*)(&s_t[lm][ks * 32 + hl * 16]);
      v8bf alo = vp[0];
      v8bf ahi = vp[1];
      v16bf a = __builtin_shufflevector(alo, ahi, 0, 1, 2, 3, 4, 5, 6, 7,
                                        8, 9, 10, 11, 12, 13, 14, 15);
      v16bf bb = *(const v16bf*)(dcw + (nt * 16 + lm) * kKd + ks * 32 + hl * 16);
      acc = __builtin_amdgcn_wmma_f32_16x16x32_bf16(false, a, false, bb,
                                                    (short)0, acc, false, false);
    }
#pragma unroll
    for (int r = 0; r < 8; ++r) {
      int m = r + hl * 8;
      s_u.part[kh][m][nt * 16 + lm] = acc[r];
    }
  }
  __syncthreads();

  // reduce the two K-halves, add bias, store (consecutive lanes -> consec. x)
  for (int i = tid; i < 16 * kO; i += 256) {
    int m = i & 15;
    int o = i >> 4;
    out[((b * kO + o) << 14) + (y << 7) + (x0 + m)] =
        s_u.part[0][m][o] + s_u.part[1][m][o] + dc_b[o];
  }
}

// ---------------------------------------------------------------------------
extern "C" void kernel_launch(void* const* d_in, const int* in_sizes, int n_in,
                              void* d_out, int out_size, void* d_ws, size_t ws_size,
                              hipStream_t stream) {
  const float* input   = (const float*)d_in[0];
  const float* offsets = (const float*)d_in[1];
  const float* com_w   = (const float*)d_in[2];
  const float* com_b   = (const float*)d_in[3];
  const float* dc_w    = (const float*)d_in[4];
  const float* dc_b    = (const float*)d_in[5];
  float* out = (float*)d_out;

  // ws layout: [comw bf16 256*576 @0][comb_pad f32 256 @294912][dcw @296960]
  __bf16* comw     = (__bf16*)d_ws;
  float*  comb_pad = (float*)((char*)d_ws + 294912);
  __bf16* dcw      = (__bf16*)((char*)d_ws + 296960);

  // 576 * 256 = 147456 threads covers comw(padded 256 rows), dcw, comb_pad
  wconv_kernel<<<576, 256, 0, stream>>>(com_w, dc_w, com_b, comw, dcw, comb_pad);

  // one workgroup per 16-pixel row chunk: 4*128*128/16 = 4096 blocks
  warpnet_fused<<<4096, 256, 0, stream>>>(input, offsets, comw, comb_pad,
                                          dcw, dc_b, out);
}
